// Layer_91113436217563
// MI455X (gfx1250) — matrix operand order repaired and checked by reading, not compiled, not run on hardware
//
#include <hip/hip_runtime.h>
#include <hip/hip_bf16.h>
#include <math.h>

// ---------------- problem constants (match reference) ----------------
#define B_   2
#define T_   2048
#define C_   1024
#define NH_  16
#define HD_  64
#define DHR_ 32
#define L_   512
#define ROWS (B_*T_)          // 4096
#define DK_  (HD_+DHR_)       // 96  (per-head q/k dim after RoPE concat)

// ---------------- CDNA5 WMMA types ----------------
typedef __attribute__((ext_vector_type(16))) _Float16 v16h;
typedef __attribute__((ext_vector_type(8)))  _Float16 v8h;
typedef __attribute__((ext_vector_type(8)))  float    v8f;

// ---------------- async global->LDS staging (gfx1250 ASYNCcnt path) -------
// The builtin expects typed vector pointers:
//   (int4 AS1* gsrc, int4 AS3* ldst, imm offset, imm cpol)
// (diagnostic from probe round: param0 = "int __attribute__((vector_size(16)))
//  __device__ *", i.e. int4 in the global address space).
#if defined(__gfx1250__)
#if __has_builtin(__builtin_amdgcn_global_load_async_to_lds_b128) && \
    __has_builtin(__builtin_amdgcn_s_wait_asynccnt)
#define HAVE_ASYNC_LDS 1
#endif
#endif

#if defined(HAVE_ASYNC_LDS)
#define AS1 __attribute__((address_space(1)))
#define AS3 __attribute__((address_space(3)))
typedef int v4i_vs __attribute__((vector_size(16)));
// copy 16 bytes global -> LDS, tracked by ASYNCcnt
#define ASYNC_CP16(gp, lp, imm)                                            \
  __builtin_amdgcn_global_load_async_to_lds_b128(                          \
      (AS1 v4i_vs*)(AS1 void*)(void*)(gp),                                 \
      (AS3 v4i_vs*)(AS3 void*)(void*)(lp), (imm), 0)
#define ASYNC_WAIT() __builtin_amdgcn_s_wait_asynccnt(0)
#endif

// Build a 16x f16 fragment from two 8-element (16B) chunks.
// Per ISA 16-bit A layout: lane<16 holds K{0..7},{16..23}; lane>=16 holds
// K{8..15},{24..31} of its row -> caller passes the right base pointers.
__device__ inline v16h frag_ld(const _Float16* p0, const _Float16* p1) {
  v8h a = *(const v8h*)p0;
  v8h b = *(const v8h*)p1;
  v16h r;
#pragma unroll
  for (int i = 0; i < 8; ++i) { r[i] = a[i]; r[i + 8] = b[i]; }
  return r;
}

__device__ inline v8f wmma_f16(v16h a, v16h b, v8f c) {
  // (neg_a, A, neg_b, B, c_mod, C, reuse_a, reuse_b)
  return __builtin_amdgcn_wmma_f32_16x16x32_f16(false, a, false, b, (short)0, c,
                                                false, false);
}

// ---------------- small helper kernels ----------------

// RMSNorm: out = x * rsqrt(mean(x^2)+eps) * g, stored f16. One block/row.
__global__ void __launch_bounds__(256)
rms_f16_kernel(const float* __restrict__ x, const float* __restrict__ g,
               _Float16* __restrict__ out, int C) {
  __shared__ float red[256];
  const int tid = threadIdx.x;
  const size_t row = blockIdx.x;
  const float* xr = x + row * C;
  float s = 0.f;
  for (int i = tid; i < C; i += 256) { float v = xr[i]; s += v * v; }
  red[tid] = s;
  __syncthreads();
  for (int o = 128; o > 0; o >>= 1) {
    if (tid < o) red[tid] += red[tid + o];
    __syncthreads();
  }
  float scale = rsqrtf(red[0] / (float)C + 1e-6f);
  for (int i = tid; i < C; i += 256)
    out[row * C + i] = (_Float16)(xr[i] * scale * g[i]);
}

// Convert + transpose weight: src f32 [K,N] -> dst f16 [N,K]
__global__ void convT_kernel(const float* __restrict__ src,
                             _Float16* __restrict__ dst, int K, int N) {
  size_t i = (size_t)blockIdx.x * 256 + threadIdx.x;
  if (i >= (size_t)K * N) return;
  int n = (int)(i % N);
  int k = (int)(i / N);
  dst[(size_t)n * K + k] = (_Float16)src[i];
}

// Interleaved-pair RoPE over [ROWS, dim]; position = (row % T) + 1.
__global__ void rope_kernel(const float* __restrict__ in,
                            float* __restrict__ out, int dim) {
  size_t i = (size_t)blockIdx.x * 256 + threadIdx.x;
  size_t total = (size_t)ROWS * (dim / 2);
  if (i >= total) return;
  int p = (int)(i % (dim / 2));
  size_t r = i / (dim / 2);
  float t = (float)(r % T_) + 1.0f;
  float theta = __expf((-2.0f * (float)p / (float)dim) * 9.210340371976184f);
  float ang = t * theta;
  float c = cosf(ang), sn = sinf(ang);
  const float* xr = in + r * dim;
  float x0 = xr[2 * p], x1 = xr[2 * p + 1];
  out[r * dim + 2 * p]     = x0 * c - x1 * sn;
  out[r * dim + 2 * p + 1] = x1 * c + x0 * sn;
}

// Pack Q: [B,NH,T,96] f16, pre-scaled by dk^-0.5. q: [B*T, C], qr: [B*T, 512]
__global__ void pack_q_kernel(const float* __restrict__ q,
                              const float* __restrict__ qr,
                              _Float16* __restrict__ Q) {
  size_t i = (size_t)blockIdx.x * 256 + threadIdx.x;
  size_t total = (size_t)B_ * NH_ * T_ * DK_;
  if (i >= total) return;
  int d = (int)(i % DK_);
  size_t r = i / DK_;
  int t = (int)(r % T_);
  size_t bh = r / T_;
  int h = (int)(bh % NH_);
  int b = (int)(bh / NH_);
  float sc = rsqrtf((float)DK_);
  float v = (d < HD_)
      ? q[((size_t)(b * T_ + t)) * C_ + h * HD_ + d]
      : qr[((size_t)(b * T_ + t)) * (NH_ * DHR_) + h * DHR_ + (d - HD_)];
  Q[i] = (_Float16)(v * sc);
}

// Pack K: [B,NH,T,96]; k = kv[:, :C] heads, kr broadcast across heads.
__global__ void pack_k_kernel(const float* __restrict__ kv,
                              const float* __restrict__ kr,
                              _Float16* __restrict__ K) {
  size_t i = (size_t)blockIdx.x * 256 + threadIdx.x;
  size_t total = (size_t)B_ * NH_ * T_ * DK_;
  if (i >= total) return;
  int d = (int)(i % DK_);
  size_t r = i / DK_;
  int t = (int)(r % T_);
  size_t bh = r / T_;
  int h = (int)(bh % NH_);
  int b = (int)(bh / NH_);
  float v = (d < HD_)
      ? kv[((size_t)(b * T_ + t)) * (2 * C_) + h * HD_ + d]
      : kr[((size_t)(b * T_ + t)) * DHR_ + (d - HD_)];
  K[i] = (_Float16)v;
}

// Pack V transposed: Vt[B,NH,64,T]; v = kv[:, C:2C] heads.
__global__ void pack_v_kernel(const float* __restrict__ kv,
                              _Float16* __restrict__ Vt) {
  size_t i = (size_t)blockIdx.x * 256 + threadIdx.x;
  size_t total = (size_t)B_ * NH_ * HD_ * T_;
  if (i >= total) return;
  int t = (int)(i % T_);
  size_t r = i / T_;
  int d = (int)(r % HD_);
  size_t bh = r / HD_;
  int h = (int)(bh % NH_);
  int b = (int)(bh / NH_);
  Vt[i] = (_Float16)kv[((size_t)(b * T_ + t)) * (2 * C_) + C_ + h * HD_ + d];
}

// ---------------- WMMA GEMM: out[M,N] = A[M,K](f16,lda) @ Bt[N,K]^T + bias ---
// modes: 0 = f32 out; 1 = f16 out; 2 = f16 out w/ tanh-gelu; 3 = f32 out + res
#define GM 128
#define GN 64
__global__ void __launch_bounds__(256)
gemm_wmma_kernel(const _Float16* __restrict__ A, int lda,
                 const _Float16* __restrict__ Bt,
                 const float* __restrict__ bias,
                 const float* __restrict__ res,
                 void* __restrict__ out,
                 int M, int N, int K, int ldc, int mode) {
  __shared__ _Float16 lA[GM * 32];
  __shared__ _Float16 lB[GN * 32];
  const int tid  = threadIdx.x;
  const int lane = tid & 31;
  const int wave = tid >> 5;           // 8 waves
  const int wm = wave & 3, wn = wave >> 2;
  const int l16 = (lane >> 4) & 1;
  const int ln  = lane & 15;
  const int m0 = blockIdx.y * GM;
  const int n0 = blockIdx.x * GN;

  v8f acc[2][2];
#pragma unroll
  for (int i = 0; i < 2; ++i)
#pragma unroll
    for (int j = 0; j < 2; ++j)
#pragma unroll
      for (int e = 0; e < 8; ++e) acc[i][j][e] = 0.f;

  const int arow = tid >> 1, ahalf = tid & 1;   // A: 128 rows x 2 halves of 16
  const int brow = tid >> 2, bq = tid & 3;      // B: 64 rows x 4 chunks of 8

  for (int k0 = 0; k0 < K; k0 += 32) {
    {  // stage A tile [128 x 32]
      const _Float16* gp = A + (size_t)(m0 + arow) * lda + k0 + ahalf * 16;
      __builtin_prefetch(gp + 32, 0, 1);          // global_prefetch_b8
      _Float16* lp = &lA[arow * 32 + ahalf * 16];
#if defined(HAVE_ASYNC_LDS)
      ASYNC_CP16(gp, lp, 0);                      // global_load_async_to_lds
      ASYNC_CP16(gp, lp, 16);
#else
      v8h a0 = *(const v8h*)gp;
      v8h a1 = *(const v8h*)(gp + 8);
      *(v8h*)lp       = a0;
      *(v8h*)(lp + 8) = a1;
#endif
    }
    {  // stage Bt tile [64 x 32] with N guard
      _Float16* lp = &lB[brow * 32 + bq * 8];
      if (n0 + brow < N) {
        const _Float16* gp = Bt + (size_t)(n0 + brow) * K + k0 + bq * 8;
#if defined(HAVE_ASYNC_LDS)
        ASYNC_CP16(gp, lp, 0);
#else
        *(v8h*)lp = *(const v8h*)gp;
#endif
      } else {
        v8h z;
#pragma unroll
        for (int i = 0; i < 8; ++i) z[i] = (_Float16)0.f;
        *(v8h*)lp = z;
      }
    }
#if defined(HAVE_ASYNC_LDS)
    ASYNC_WAIT();                                 // s_wait_asynccnt 0
#endif
    __syncthreads();

    v16h af[2], bf[2];
#pragma unroll
    for (int i = 0; i < 2; ++i) {
      const _Float16* ap = &lA[(wm * 32 + i * 16 + ln) * 32 + l16 * 8];
      af[i] = frag_ld(ap, ap + 16);
    }
#pragma unroll
    for (int j = 0; j < 2; ++j) {
      const _Float16* bp = &lB[(wn * 32 + j * 16 + ln) * 32 + l16 * 8];
      bf[j] = frag_ld(bp, bp + 16);
    }
#pragma unroll
    for (int i = 0; i < 2; ++i)
#pragma unroll
      for (int j = 0; j < 2; ++j)
        acc[i][j] = wmma_f16(af[i], bf[j], acc[i][j]);
    __syncthreads();
  }

  // epilogue (D layout: VGPR e -> row e + 8*(lane>=16), col = lane&15)
#pragma unroll
  for (int i = 0; i < 2; ++i)
#pragma unroll
    for (int j = 0; j < 2; ++j)
#pragma unroll
      for (int e = 0; e < 8; ++e) {
        int row = m0 + wm * 32 + i * 16 + e + 8 * l16;
        int col = n0 + wn * 32 + j * 16 + ln;
        if (row < M && col < N) {
          float v = acc[i][j][e] + (bias ? bias[col] : 0.f);
          size_t off = (size_t)row * ldc + col;
          if (mode == 0) {
            ((float*)out)[off] = v;
          } else if (mode == 1) {
            ((_Float16*)out)[off] = (_Float16)v;
          } else if (mode == 2) {
            float gl = 0.5f * v *
                (1.f + tanhf(0.7978845608028654f * (v + 0.044715f * v * v * v)));
            ((_Float16*)out)[off] = (_Float16)gl;
          } else {
            ((float*)out)[off] = v + res[off];
          }
        }
      }
}

// ---------------- WMMA flash attention ----------------
// grid = B*NH*(T/64); block = 128 (4 waves); wave handles 16 query rows.
// Q,K: [B*NH, T, 96] f16 (Q pre-scaled); Vt: [B*NH, 64, T] f16.
// O: [B*T, C] f16 (head-interleaved layout for the wo GEMM).
__global__ void __launch_bounds__(128)
attn_wmma_kernel(const _Float16* __restrict__ Q,
                 const _Float16* __restrict__ Kc,
                 const _Float16* __restrict__ Vt,
                 _Float16* __restrict__ O) {
  __shared__ _Float16 sK[32 * 96];       // key block [32 keys][96 dims]
  __shared__ _Float16 sV[64 * 32];       // V^T block [64 dims][32 keys]
  __shared__ _Float16 sP[4 * 16 * 32];   // per-wave probability tile

  const int tid = threadIdx.x;
  const int lane = tid & 31, wave = tid >> 5;
  const int l16 = (lane >> 4) & 1, ln = lane & 15;
  const int QB = T_ / 64;                // 32 query blocks per (b,h)
  const int qb = blockIdx.x % QB;
  const int bh = blockIdx.x / QB;
  const int b = bh / NH_, h = bh % NH_;
  const int q0 = qb * 64 + wave * 16;

  const _Float16* Qg = Q  + (size_t)bh * T_ * DK_;
  const _Float16* Kg = Kc + (size_t)bh * T_ * DK_;
  const _Float16* Vg = Vt + (size_t)bh * HD_ * T_;

  // preload 3 Q fragments (K-dim 96 = 3x32) straight from global
  v16h qf[3];
#pragma unroll
  for (int kc = 0; kc < 3; ++kc) {
    const _Float16* p = Qg + (size_t)(q0 + ln) * DK_ + kc * 32 + l16 * 8;
    qf[kc] = frag_ld(p, p + 16);
  }

  v8f oacc[4];
  float mrow[8], lrow[8];
#pragma unroll
  for (int n = 0; n < 4; ++n)
#pragma unroll
    for (int e = 0; e < 8; ++e) oacc[n][e] = 0.f;
#pragma unroll
  for (int e = 0; e < 8; ++e) { mrow[e] = -1e30f; lrow[e] = 0.f; }

  const int numKB = (qb * 64 + 64) / 32;  // causal extent for this block
  for (int kb = 0; kb < numKB; ++kb) {
    const int kbase = kb * 32;
    {  // stage K block: 32x96, 128 threads x 24 elems (3x16B)
      int r = tid >> 2, seg = tid & 3;
      const _Float16* gp = Kg + (size_t)(kbase + r) * DK_ + seg * 24;
      _Float16* lp = &sK[r * 96 + seg * 24];
#if defined(HAVE_ASYNC_LDS)
      ASYNC_CP16(gp, lp, 0);
      ASYNC_CP16(gp, lp, 16);
      ASYNC_CP16(gp, lp, 32);
#else
      *(v8h*)lp        = *(const v8h*)gp;
      *(v8h*)(lp + 8)  = *(const v8h*)(gp + 8);
      *(v8h*)(lp + 16) = *(const v8h*)(gp + 16);
#endif
    }
    {  // stage V^T block: 64x32, 128 threads x 16 elems (2x16B)
      int d = tid >> 1, hh = tid & 1;
      const _Float16* gp = Vg + (size_t)d * T_ + kbase + hh * 16;
      _Float16* lp = &sV[d * 32 + hh * 16];
#if defined(HAVE_ASYNC_LDS)
      ASYNC_CP16(gp, lp, 0);
      ASYNC_CP16(gp, lp, 16);
#else
      *(v8h*)lp       = *(const v8h*)gp;
      *(v8h*)(lp + 8) = *(const v8h*)(gp + 8);
#endif
    }
#if defined(HAVE_ASYNC_LDS)
    ASYNC_WAIT();
#endif
    __syncthreads();

    // S tile = Q @ K^T  (16 rows x 32 keys, 2 n-tiles x 3 k-chunks)
    v8f s[2];
#pragma unroll
    for (int j = 0; j < 2; ++j)
#pragma unroll
      for (int e = 0; e < 8; ++e) s[j][e] = 0.f;
#pragma unroll
    for (int kc = 0; kc < 3; ++kc)
#pragma unroll
      for (int j = 0; j < 2; ++j) {
        const _Float16* bp = &sK[(j * 16 + ln) * 96 + kc * 32 + l16 * 8];
        s[j] = wmma_f16(qf[kc], frag_ld(bp, bp + 16), s[j]);
      }

    // causal mask + online softmax (rows live in D layout: e + 8*l16)
#pragma unroll
    for (int e = 0; e < 8; ++e) {
      int qi = q0 + e + 8 * l16;
      int c0 = kbase + ln, c1 = kbase + 16 + ln;
      float v0 = (c0 <= qi) ? s[0][e] : -1e30f;
      float v1 = (c1 <= qi) ? s[1][e] : -1e30f;
      float rm = fmaxf(v0, v1);
#pragma unroll
      for (int msk = 1; msk < 16; msk <<= 1)
        rm = fmaxf(rm, __shfl_xor(rm, msk, 32));
      float mn = fmaxf(mrow[e], rm);
      float alpha = __expf(mrow[e] - mn);
      v0 = __expf(v0 - mn);
      v1 = __expf(v1 - mn);
      float rs = v0 + v1;
#pragma unroll
      for (int msk = 1; msk < 16; msk <<= 1) rs += __shfl_xor(rs, msk, 32);
      lrow[e] = lrow[e] * alpha + rs;
      mrow[e] = mn;
#pragma unroll
      for (int n = 0; n < 4; ++n) oacc[n][e] *= alpha;
      s[0][e] = v0;
      s[1][e] = v1;
    }

    // spill P (D layout) -> LDS, reload as A-layout fragment
    _Float16* Pw = &sP[wave * 16 * 32];
#pragma unroll
    for (int e = 0; e < 8; ++e) {
      int r = e + 8 * l16;
      Pw[r * 32 + ln]      = (_Float16)s[0][e];
      Pw[r * 32 + 16 + ln] = (_Float16)s[1][e];
    }
    __syncthreads();

    const _Float16* pp = &sP[wave * 16 * 32 + ln * 32 + l16 * 8];
    v16h pf = frag_ld(pp, pp + 16);
#pragma unroll
    for (int n = 0; n < 4; ++n) {
      const _Float16* vp = &sV[(n * 16 + ln) * 32 + l16 * 8];
      oacc[n] = wmma_f16(pf, frag_ld(vp, vp + 16), oacc[n]);
    }
    __syncthreads();
  }

  // normalize and store O as [B*T, C] f16 (head-interleaved)
#pragma unroll
  for (int n = 0; n < 4; ++n)
#pragma unroll
    for (int e = 0; e < 8; ++e) {
      int qi = q0 + e + 8 * l16;
      float v = oacc[n][e] / lrow[e];
      O[((size_t)(b * T_ + qi)) * C_ + h * HD_ + n * 16 + ln] = (_Float16)v;
    }
}

// ---------------- host orchestration ----------------
extern "C" void kernel_launch(void* const* d_in, const int* in_sizes, int n_in,
                              void* d_out, int out_size, void* d_ws,
                              size_t ws_size, hipStream_t stream) {
  (void)in_sizes; (void)n_in; (void)out_size; (void)ws_size;
  const float* x   = (const float*)d_in[0];
  const float* w1  = (const float*)d_in[1];
  const float* b1  = (const float*)d_in[2];
  const float* wkr = (const float*)d_in[3];
  const float* bkr = (const float*)d_in[4];
  const float* wqr = (const float*)d_in[5];
  const float* bqr = (const float*)d_in[6];
  const float* wkv = (const float*)d_in[7];
  const float* bkv = (const float*)d_in[8];
  const float* wq  = (const float*)d_in[9];
  const float* bq  = (const float*)d_in[10];
  const float* wo  = (const float*)d_in[11];
  const float* bo  = (const float*)d_in[12];
  const float* wi  = (const float*)d_in[13];
  const float* bi  = (const float*)d_in[14];
  const float* wo2 = (const float*)d_in[15];
  const float* bo2 = (const float*)d_in[16];
  const float* g1  = (const float*)d_in[17];
  const float* g2  = (const float*)d_in[18];

  char* ws = (char*)d_ws;
  size_t off = 0;
  auto alloc = [&](size_t bytes) -> void* {
    void* p = ws + off;
    off += (bytes + 255) & ~((size_t)255);
    return p;
  };

  // f16 transposed weights [N,K]
  _Float16* w1t  = (_Float16*)alloc((size_t)1024 * 1024 * 2);
  _Float16* wkrt = (_Float16*)alloc((size_t)32   * 1024 * 2);
  _Float16* wqrt = (_Float16*)alloc((size_t)512  * 1024 * 2);
  _Float16* wkvt = (_Float16*)alloc((size_t)2048 * 512  * 2);
  _Float16* wqt  = (_Float16*)alloc((size_t)1024 * 512  * 2);
  _Float16* wot  = (_Float16*)alloc((size_t)1024 * 1024 * 2);
  _Float16* wit  = (_Float16*)alloc((size_t)4096 * 1024 * 2);
  _Float16* wo2t = (_Float16*)alloc((size_t)1024 * 4096 * 2);
  // activations
  _Float16* xn   = (_Float16*)alloc((size_t)ROWS * 1024 * 2);
  _Float16* hb   = (_Float16*)alloc((size_t)ROWS * 1024 * 2);
  float*    krp  = (float*)   alloc((size_t)ROWS * 32   * 4);
  float*    krr  = (float*)   alloc((size_t)ROWS * 32   * 4);
  float*    qrp  = (float*)   alloc((size_t)ROWS * 512  * 4);
  float*    qrr  = (float*)   alloc((size_t)ROWS * 512  * 4);
  float*    kvf  = (float*)   alloc((size_t)ROWS * 2048 * 4);
  float*    qfv  = (float*)   alloc((size_t)ROWS * 1024 * 4);
  _Float16* Qp   = (_Float16*)alloc((size_t)B_ * NH_ * T_ * DK_ * 2);
  _Float16* Kp   = (_Float16*)alloc((size_t)B_ * NH_ * T_ * DK_ * 2);
  _Float16* Vtp  = (_Float16*)alloc((size_t)B_ * NH_ * HD_ * T_ * 2);
  _Float16* Obf  = (_Float16*)alloc((size_t)ROWS * 1024 * 2);
  float*    x2   = (float*)   alloc((size_t)ROWS * 1024 * 4);
  _Float16* xn2  = (_Float16*)alloc((size_t)ROWS * 1024 * 2);
  _Float16* ff1  = (_Float16*)alloc((size_t)ROWS * 4096 * 2);

  auto convT = [&](const float* s, _Float16* d, int K, int N) {
    size_t tot = (size_t)K * N;
    convT_kernel<<<(unsigned)((tot + 255) / 256), 256, 0, stream>>>(s, d, K, N);
  };
  auto gemm = [&](const _Float16* A, int lda, const _Float16* Bt,
                  const float* bias, const float* res, void* out,
                  int M, int N, int K, int ldc, int mode) {
    dim3 grid((N + GN - 1) / GN, M / GM);
    gemm_wmma_kernel<<<grid, 256, 0, stream>>>(A, lda, Bt, bias, res, out, M, N,
                                               K, ldc, mode);
  };

  // weight conversion + transpose
  convT(w1,  w1t,  1024, 1024);
  convT(wkr, wkrt, 1024, 32);
  convT(wqr, wqrt, 1024, 512);
  convT(wkv, wkvt, 512,  2048);
  convT(wq,  wqt,  512,  1024);
  convT(wo,  wot,  1024, 1024);
  convT(wi,  wit,  1024, 4096);
  convT(wo2, wo2t, 4096, 1024);

  // xn = rms(x, g1)
  rms_f16_kernel<<<ROWS, 256, 0, stream>>>(x, g1, xn, C_);
  // h = xn @ w1 + b1  (f16 out)
  gemm(xn, 1024, w1t, b1, nullptr, hb, ROWS, 1024, 1024, 1024, 1);
  // kr_pre = h @ wkr + bkr ; qr_pre = h @ wqr + bqr  (f32 out)
  gemm(hb, 1024, wkrt, bkr, nullptr, krp, ROWS, 32, 1024, 32, 0);
  gemm(hb, 1024, wqrt, bqr, nullptr, qrp, ROWS, 512, 1024, 512, 0);
  // RoPE
  {
    size_t t1 = (size_t)ROWS * (32 / 2);
    rope_kernel<<<(unsigned)((t1 + 255) / 256), 256, 0, stream>>>(krp, krr, 32);
    size_t t2 = (size_t)ROWS * (512 / 2);
    rope_kernel<<<(unsigned)((t2 + 255) / 256), 256, 0, stream>>>(qrp, qrr, 512);
  }
  // kv = kv_lat @ wkv + bkv  (kv_lat = h[:, :512]); q = q_lat @ wq + bq
  gemm(hb, 1024, wkvt, bkv, nullptr, kvf, ROWS, 2048, 512, 2048, 0);
  gemm(hb + 512, 1024, wqt, bq, nullptr, qfv, ROWS, 1024, 512, 1024, 0);
  // pack Q/K/Vt
  {
    size_t tq = (size_t)B_ * NH_ * T_ * DK_;
    pack_q_kernel<<<(unsigned)((tq + 255) / 256), 256, 0, stream>>>(qfv, qrr, Qp);
    pack_k_kernel<<<(unsigned)((tq + 255) / 256), 256, 0, stream>>>(kvf, krr, Kp);
    size_t tv = (size_t)B_ * NH_ * HD_ * T_;
    pack_v_kernel<<<(unsigned)((tv + 255) / 256), 256, 0, stream>>>(kvf, Vtp);
  }
  // flash attention
  attn_wmma_kernel<<<B_ * NH_ * (T_ / 64), 128, 0, stream>>>(Qp, Kp, Vtp, Obf);
  // x2 = x + O @ wo + bo
  gemm(Obf, 1024, wot, bo, x, x2, ROWS, 1024, 1024, 1024, 3);
  // FFN
  rms_f16_kernel<<<ROWS, 256, 0, stream>>>(x2, g2, xn2, C_);
  gemm(xn2, 1024, wit, bi, nullptr, ff1, ROWS, 4096, 1024, 4096, 2);
  gemm(ff1, 4096, wo2t, bo2, x2, d_out, ROWS, 1024, 4096, 1024, 3);
}
